// NeuralCausalDiscovery_45732811768456
// MI455X (gfx1250) — matrix operand order repaired and checked by reading, not compiled, not run on hardware
//
#include <hip/hip_runtime.h>
#include <hip/hip_bf16.h>

#define BB 16
#define NN 128
#define HH 256
#define NHEADS 8
#define DHEAD 32

typedef __bf16 bfloat_t;
typedef bfloat_t v16bf __attribute__((ext_vector_type(16)));
typedef bfloat_t v8bf  __attribute__((ext_vector_type(8)));
typedef bfloat_t v4bf  __attribute__((ext_vector_type(4)));
typedef float    v8f   __attribute__((ext_vector_type(8)));

#define SHUF16(lo, hi) __builtin_shufflevector((lo), (hi), 0,1,2,3,4,5,6,7,8,9,10,11,12,13,14,15)

// ---------- helpers ----------

__device__ __forceinline__ bfloat_t f2bf(float x) { return (bfloat_t)x; }  // RNE

__device__ __forceinline__ v8f wmma_bf16(v16bf a, v16bf b, v8f c) {
  // v_wmma_f32_16x16x32_bf16  (neg_a, A, neg_b, B, c_mod, C, reuse_a, reuse_b)
  return __builtin_amdgcn_wmma_f32_16x16x32_bf16(false, a, false, b, (short)0, c,
                                                 false, false);
}

// A tile 16x32 bf16, row-major source: lane(m=L&15,hf=L>>4) elems e0..15 -> k=hf*16+e:
// one contiguous 16-element run (two 16B loads).
__device__ __forceinline__ v16bf load_a_chunks(const bfloat_t* __restrict__ p) {
  int hf = (threadIdx.x & 31) >> 4;
  const v8bf* pa = (const v8bf*)p;            // p already at row + k0 + hf*8; run starts at row + k0 + hf*16
  return SHUF16(pa[hf], pa[hf + 1]);
}

// Strided bf16 B gather (used only for attention's V matrix): b[e]=Bp[(hf*16+e)*ldb+n]
__device__ __forceinline__ v16bf load_b_gather_bf(const bfloat_t* __restrict__ Bp, int ldb) {
  int lane = threadIdx.x & 31;
  int n = lane & 15, hf = lane >> 4;
  v16bf b;
#pragma unroll
  for (int e = 0; e < 16; ++e) b[e] = Bp[(size_t)(hf * 16 + e) * ldb + n];
  return b;
}

// ---------- weight packing: f32 [K,N] -> bf16 WMMA-B layout ----------
// Wp[((kt*(N/16)+nt)*32 + lane)*16 + e] = W[kt*32 + (lane>>4)*16 + e][nt*16 + (lane&15)]
__global__ void ncd_pack_w_kernel(const float* __restrict__ W, bfloat_t* __restrict__ Wp,
                                  int K, int N) {
  int idx = blockIdx.x * 256 + threadIdx.x;
  if (idx >= K * N) return;
  int e = idx & 15;
  int lane = (idx >> 4) & 31;
  int rest = idx >> 9;
  int nt16 = N >> 4;
  int nt = rest % nt16, kt = rest / nt16;
  int k = kt * 32 + ((lane >> 4) << 4) + e;
  int n = nt * 16 + (lane & 15);
  Wp[idx] = f2bf(W[(size_t)k * N + n]);
}

// ---------- elementwise ----------

__global__ void ncd_prep_kernel(const float* __restrict__ vd, const float* __restrict__ emb,
                                float* __restrict__ h0f, bfloat_t* __restrict__ h0b) {
  int idx = blockIdx.x * 256 + threadIdx.x;
  if (idx < BB * NN * HH) {
    float v = vd[idx] + emb[idx & (NN * HH - 1)];
    h0f[idx] = v;
    h0b[idx] = f2bf(v);
  }
}

// ---------- GEMM: bf16 A (row-major) x bf16 packed B -> f32/bf16 C ----------
// Wave computes a 16x64 output strip (4 accumulators sharing one A tile).
__global__ void ncd_gemm_bf_kernel(const bfloat_t* __restrict__ A, const bfloat_t* __restrict__ Bp,
                                   const float* __restrict__ bias, const float* __restrict__ resid,
                                   float* __restrict__ Cf, bfloat_t* __restrict__ Cb,
                                   int M, int N, int K, int do_relu) {
  int ncol4 = N >> 6;
  int wt = (M >> 4) * ncol4;
  int tile = blockIdx.x * 8 + (threadIdx.x >> 5);
  if (tile >= wt) return;  // wave-uniform
  int tm = tile / ncol4, tg = tile - tm * ncol4;
  int lane = threadIdx.x & 31;
  int m = lane & 15, hf = lane >> 4, n = m, mb = hf << 3;
  int nt16 = N >> 4;
  v8f acc[4] = {};
  const bfloat_t* Arow = A + (size_t)(tm * 16 + m) * K + hf * 8;
#pragma unroll 4
  for (int k0 = 0; k0 < K; k0 += 32) {
    __builtin_prefetch(Arow + k0 + 128, 0, 0);        // global_prefetch_b8
    v16bf a = load_a_chunks(Arow + k0);
    int kt = k0 >> 5;
#pragma unroll
    for (int c = 0; c < 4; ++c) {
      const v16bf* pb = (const v16bf*)(Bp + (((size_t)kt * nt16 + (tg * 4 + c)) * 32 + lane) * 16);
      acc[c] = wmma_bf16(a, *pb, acc[c]);
    }
  }
#pragma unroll
  for (int c = 0; c < 4; ++c) {
    int col = (tg * 4 + c) * 16 + n;
    float bv = bias ? bias[col] : 0.0f;
#pragma unroll
    for (int r = 0; r < 8; ++r) {
      int row = tm * 16 + mb + r;
      float v = acc[c][r] + bv;
      if (resid) v += resid[(size_t)row * N + col];
      if (do_relu) v = fmaxf(v, 0.0f);
      if (Cf) Cf[(size_t)row * N + col] = v;
      if (Cb) Cb[(size_t)row * N + col] = f2bf(v);
    }
  }
}

// ---------- attention: one block per (b, head) ----------
__global__ void ncd_attn_kernel(const bfloat_t* __restrict__ qkvb,
                                float* __restrict__ attn_all,
                                bfloat_t* __restrict__ ctxb) {
  __shared__ __align__(32) float S[NN * NN];      // 64 KB
  __shared__ __align__(32) bfloat_t Sb[NN * NN];  // 32 KB
  int bh = blockIdx.x;
  int b = bh >> 3, hd = bh & 7;
  int tid = threadIdx.x;
  int wave = tid >> 5, lane = tid & 31;
  int m = lane & 15, n = lane & 15, hf = lane >> 4, mb = hf << 3;
  const bfloat_t* qb = qkvb + (size_t)(b * NN) * (3 * HH) + hd * DHEAD;
  const bfloat_t* kb = qb + HH;
  const bfloat_t* vb = qb + 2 * HH;

  // scores = q @ k^T / sqrt(DH): 8x8 tiles, K=32 -> one WMMA each.
  // k^T B-tile = 16 contiguous bf16 per lane (row of k), single 32B load.
  for (int t = wave; t < 64; t += 8) {
    int tm = t >> 3, tn = t & 7;
    v16bf a = load_a_chunks(qb + (size_t)(tm * 16 + m) * (3 * HH) + hf * 8);
    const v16bf* pb = (const v16bf*)(kb + (size_t)(tn * 16 + n) * (3 * HH) + hf * 16);
    v8f acc = {};
    acc = wmma_bf16(a, *pb, acc);
#pragma unroll
    for (int r = 0; r < 8; ++r)
      S[(tm * 16 + mb + r) * NN + tn * 16 + n] = acc[r] * 0.17677669529663687f;
  }
  __syncthreads();

  // row softmax (f32) + bf16 copy for the ctx GEMM
  if (tid < NN) {
    float* row = S + tid * NN;
    bfloat_t* rowb = Sb + tid * NN;
    float mx = -1e30f;
    for (int j = 0; j < NN; ++j) mx = fmaxf(mx, row[j]);
    float sum = 0.0f;
    for (int j = 0; j < NN; ++j) { float e = __expf(row[j] - mx); row[j] = e; sum += e; }
    float inv = 1.0f / sum;
    for (int j = 0; j < NN; ++j) { float v = row[j] * inv; row[j] = v; rowb[j] = f2bf(v); }
  }
  __syncthreads();

  float* ao = attn_all + (size_t)bh * (NN * NN);
  for (int idx = tid; idx < NN * NN; idx += 256) ao[idx] = S[idx];

  // ctx = attn @ v : 8x2 tiles, K=128 from LDS bf16
  for (int t = wave; t < 16; t += 8) {
    int tm = t >> 1, tn = t & 1;
    v8f acc = {};
#pragma unroll
    for (int kt = 0; kt < 4; ++kt) {
      v16bf a = load_a_chunks(Sb + (size_t)(tm * 16 + m) * NN + kt * 32 + hf * 8);
      v16bf bmat = load_b_gather_bf(vb + (size_t)kt * 32 * (3 * HH) + tn * 16, 3 * HH);
      acc = wmma_bf16(a, bmat, acc);
    }
#pragma unroll
    for (int r = 0; r < 8; ++r)
      ctxb[(size_t)(b * NN + tm * 16 + mb + r) * HH + hd * DHEAD + tn * 16 + n] = f2bf(acc[r]);
  }
}

__global__ void ncd_attn_mean_kernel(const float* __restrict__ attn_all,
                                     float* __restrict__ out) {
  int idx = blockIdx.x * 256 + threadIdx.x;
  if (idx >= BB * NN * NN) return;
  int b = idx >> 14, rem = idx & 16383;
  float s = 0.0f;
#pragma unroll
  for (int hd = 0; hd < NHEADS; ++hd)
    s += attn_all[((size_t)(b * NHEADS + hd) << 14) + rem];
  out[idx] = s * 0.125f;
}

__global__ void ncd_ln_kernel(const float* __restrict__ x, const float* __restrict__ g,
                              const float* __restrict__ bb, bfloat_t* __restrict__ y) {
  __shared__ float red[256];
  int row = blockIdx.x, t = threadIdx.x;
  float v = x[(size_t)row * HH + t];
  red[t] = v;
  __syncthreads();
  for (int s = 128; s > 0; s >>= 1) { if (t < s) red[t] += red[t + s]; __syncthreads(); }
  float mean = red[0] * (1.0f / HH);
  __syncthreads();
  float d = v - mean;
  red[t] = d * d;
  __syncthreads();
  for (int s = 128; s > 0; s >>= 1) { if (t < s) red[t] += red[t + s]; __syncthreads(); }
  float var = red[0] * (1.0f / HH);
  y[(size_t)row * HH + t] = f2bf(d * rsqrtf(var + 1e-5f) * g[t] + bb[t]);
}

__global__ void ncd_copy_kernel(const float* __restrict__ src, float* __restrict__ dst, int n) {
  int idx = blockIdx.x * 256 + threadIdx.x;
  if (idx < n) dst[idx] = src[idx];
}

__global__ void ncd_kl_part_kernel(const float* __restrict__ mu, const float* __restrict__ lv,
                                   float* __restrict__ part) {
  __shared__ float red[256];
  int idx = blockIdx.x * 256 + threadIdx.x;
  float m = mu[idx], l = lv[idx];
  red[threadIdx.x] = 1.0f + l - m * m - __expf(l);
  __syncthreads();
  for (int s = 128; s > 0; s >>= 1) {
    if (threadIdx.x < s) red[threadIdx.x] += red[threadIdx.x + s];
    __syncthreads();
  }
  if (threadIdx.x == 0) part[blockIdx.x] = red[0];
}

__global__ void ncd_kl_final_kernel(const float* __restrict__ part, float* __restrict__ kl) {
  __shared__ float red[256];
  float s = 0.0f;
  for (int i = threadIdx.x; i < 2048; i += 256) s += part[i];
  red[threadIdx.x] = s;
  __syncthreads();
  for (int st = 128; st > 0; st >>= 1) {
    if (threadIdx.x < st) red[threadIdx.x] += red[threadIdx.x + st];
    __syncthreads();
  }
  if (threadIdx.x == 0) *kl = red[0] * (-0.5f / (float)BB);
}

// ---------- fused pairwise scoring: one block per (b,i) ----------
__global__ void ncd_pairwise_kernel(
    const float* __restrict__ a_cp, const float* __restrict__ c_cp,
    const float* __restrict__ a_se, const float* __restrict__ c_se,
    const float* __restrict__ a_pc, const float* __restrict__ c_pc,
    const bfloat_t* __restrict__ cpW2P, const float* __restrict__ cpb2,
    const float* __restrict__ cpW3, const float* __restrict__ cpb3,
    const float* __restrict__ seW2, const float* __restrict__ seb2,
    const float* __restrict__ pcW2, const float* __restrict__ pcb2,
    float* __restrict__ out_cp, float* __restrict__ out_se, float* __restrict__ out_pc) {
  __shared__ __align__(16) bfloat_t Z[64 * HH];   // 32 KB (half of the j-rows)
  __shared__ __align__(16) float ai_cp[HH], ai_se[HH], ai_pc[HH];
  __shared__ float wacc[8][NN];                   // per-wave partials (deterministic)
  __shared__ float part[256];

  int bi = blockIdx.x;
  int b = bi >> 7, i = bi & 127;
  int tid = threadIdx.x;
  int wave = tid >> 5, lane = tid & 31;
  int m = lane & 15, n = lane & 15, hf = lane >> 4, mb = hf << 3;
  size_t rowi = (size_t)(b * NN + i) * HH;

  ai_cp[tid] = a_cp[rowi + tid];  // b1 already folded into a_*
  ai_se[tid] = a_se[rowi + tid];
  ai_pc[tid] = a_pc[rowi + tid];
  __syncthreads();

  // ---- cp head: Z(128x256) @ W2(256x128) -> relu(+b2) -> dot W3 ----
  for (int jh = 0; jh < 2; ++jh) {
    __syncthreads();
    {  // build Z half: thread owns a contiguous 64-element quarter-row
      int j = tid >> 2;
      int kb = (tid & 3) << 6;
      const float* crow = c_cp + (size_t)(b * NN + (jh << 6) + j) * HH + kb;
      const float* arow = ai_cp + kb;
      v4bf* zrow = (v4bf*)(Z + j * HH + kb);
#pragma unroll
      for (int q = 0; q < 16; ++q) {
        float4 cv = ((const float4*)crow)[q];
        float4 av = ((const float4*)arow)[q];
        v4bf zz;
        zz[0] = f2bf(fmaxf(av.x + cv.x, 0.0f));
        zz[1] = f2bf(fmaxf(av.y + cv.y, 0.0f));
        zz[2] = f2bf(fmaxf(av.z + cv.z, 0.0f));
        zz[3] = f2bf(fmaxf(av.w + cv.w, 0.0f));
        zrow[q] = zz;
      }
    }
    __syncthreads();
    // wave owns one 16-column tile of W2 (tn = wave); B tile reused over 4 M-tiles
    int tn = wave;
    v8f acc[4] = {};
#pragma unroll
    for (int kt = 0; kt < 8; ++kt) {
      const v16bf* pb = (const v16bf*)(cpW2P + (((size_t)kt * 8 + tn) * 32 + lane) * 16);
      v16bf bmat = *pb;
#pragma unroll
      for (int tm = 0; tm < 4; ++tm) {
        v16bf a = load_a_chunks(Z + (size_t)(tm * 16 + m) * HH + kt * 32 + hf * 8);
        acc[tm] = wmma_bf16(a, bmat, acc[tm]);
      }
    }
    int col = tn * 16 + n;
    float w3 = cpW3[col], b2v = cpb2[col];
#pragma unroll
    for (int tm = 0; tm < 4; ++tm) {
#pragma unroll
      for (int r = 0; r < 8; ++r) {
        float s = fmaxf(acc[tm][r] + b2v, 0.0f) * w3;
        s += __shfl_xor(s, 1, 32);
        s += __shfl_xor(s, 2, 32);
        s += __shfl_xor(s, 4, 32);
        s += __shfl_xor(s, 8, 32);
        if (n == 0) wacc[wave][(jh << 6) + tm * 16 + mb + r] = s;
      }
    }
  }
  __syncthreads();
  if (tid < NN) {
    float s = cpb3[0];
#pragma unroll
    for (int w = 0; w < 8; ++w) s += wacc[w][tid];
    float v = 1.0f / (1.0f + __expf(-s));
    out_cp[(size_t)(b * NN + i) * NN + tid] = (tid == i) ? 0.0f : v;
  }

  // ---- se head: tanh(relu(a_i + c_j) . W2 + b2) ----
  {
    int j = tid >> 1, off = (tid & 1) << 7;
    const float* crow = c_se + (size_t)(b * NN + j) * HH + off;
    const float* arow = ai_se + off;
    const float* wrow = seW2 + off;
    float s = 0.0f;
#pragma unroll
    for (int q = 0; q < 32; ++q) {
      float4 cv = ((const float4*)crow)[q];
      float4 av = ((const float4*)arow)[q];
      float4 wv = ((const float4*)wrow)[q];
      s += fmaxf(av.x + cv.x, 0.0f) * wv.x + fmaxf(av.y + cv.y, 0.0f) * wv.y +
           fmaxf(av.z + cv.z, 0.0f) * wv.z + fmaxf(av.w + cv.w, 0.0f) * wv.w;
    }
    part[tid] = s;
  }
  __syncthreads();
  if (tid < NN) {
    float v = tanhf(part[2 * tid] + part[2 * tid + 1] + seb2[0]);
    out_se[(size_t)(b * NN + i) * NN + tid] = (tid == i) ? 0.0f : v;
  }
  __syncthreads();

  // ---- pc head: sigmoid(relu(a_i + c_j) . W2 + b2) ----
  {
    int j = tid >> 1, off = (tid & 1) << 7;
    const float* crow = c_pc + (size_t)(b * NN + j) * HH + off;
    const float* arow = ai_pc + off;
    const float* wrow = pcW2 + off;
    float s = 0.0f;
#pragma unroll
    for (int q = 0; q < 32; ++q) {
      float4 cv = ((const float4*)crow)[q];
      float4 av = ((const float4*)arow)[q];
      float4 wv = ((const float4*)wrow)[q];
      s += fmaxf(av.x + cv.x, 0.0f) * wv.x + fmaxf(av.y + cv.y, 0.0f) * wv.y +
           fmaxf(av.z + cv.z, 0.0f) * wv.z + fmaxf(av.w + cv.w, 0.0f) * wv.w;
    }
    part[tid] = s;
  }
  __syncthreads();
  if (tid < NN) {
    float s = part[2 * tid] + part[2 * tid + 1] + pcb2[0];
    float v = 1.0f / (1.0f + __expf(-s));
    out_pc[(size_t)(b * NN + i) * NN + tid] = (tid == i) ? 0.0f : v;
  }
}

// ---------- launch ----------

extern "C" void kernel_launch(void* const* d_in, const int* in_sizes, int n_in,
                              void* d_out, int out_size, void* d_ws, size_t ws_size,
                              hipStream_t stream) {
  (void)in_sizes; (void)n_in; (void)out_size; (void)ws_size;
  const float* vd   = (const float*)d_in[0];
  const float* emb  = (const float*)d_in[1];
  const float* Wqkv = (const float*)d_in[2];
  const float* bqkv = (const float*)d_in[3];
  const float* Wo   = (const float*)d_in[4];
  const float* bo   = (const float*)d_in[5];
  const float* ln_g = (const float*)d_in[6];
  const float* ln_b = (const float*)d_in[7];
  const float* gnnW = (const float*)d_in[8];
  const float* gnnb = (const float*)d_in[9];
  const float* muW  = (const float*)d_in[10];
  const float* mub  = (const float*)d_in[11];
  const float* lvW  = (const float*)d_in[12];
  const float* lvb  = (const float*)d_in[13];
  const float* cpW1 = (const float*)d_in[14];
  const float* cpb1 = (const float*)d_in[15];
  const float* cpW2 = (const float*)d_in[16];
  const float* cpb2 = (const float*)d_in[17];
  const float* cpW3 = (const float*)d_in[18];
  const float* cpb3 = (const float*)d_in[19];
  const float* seW1 = (const float*)d_in[20];
  const float* seb1 = (const float*)d_in[21];
  const float* seW2 = (const float*)d_in[22];
  const float* seb2 = (const float*)d_in[23];
  const float* pcW1 = (const float*)d_in[24];
  const float* pcb1 = (const float*)d_in[25];
  const float* pcW2 = (const float*)d_in[26];
  const float* pcb2 = (const float*)d_in[27];

  float* out     = (float*)d_out;
  float* o_cp    = out;
  float* o_se    = out + 262144;
  float* o_pc    = out + 524288;
  float* o_hvar  = out + 786432;
  float* o_attnw = out + 1310720;
  float* o_mu    = out + 1572864;
  float* o_lv    = out + 2097152;
  float* o_kl    = out + 2621440;

  char* base = (char*)d_ws;
  const size_t MB = 1024 * 1024;
  // f32 region (with overlays; lifetimes are disjoint)
  float* h0f      = (float*)(base + 0 * MB);    // dead after xres GEMM
  float* attn_all = (float*)(base + 2 * MB);    // 8MB, dead after head-mean
  float* xres     = (float*)(base + 10 * MB);   // dead after LN
  float* a_cp     = (float*)(base + 0 * MB);    // overlays h0f
  float* c_cp     = (float*)(base + 2 * MB);    // overlays attn_all
  float* a_se     = (float*)(base + 4 * MB);
  float* c_se     = (float*)(base + 6 * MB);
  float* a_pc     = (float*)(base + 8 * MB);
  float* c_pc     = (float*)(base + 10 * MB);   // overlays xres
  float* klp      = (float*)(base + 12 * MB);   // 8KB
  // bf16 region
  char* p = base + 12 * MB + 8192;
  bfloat_t* h0b    = (bfloat_t*)p; p += 1048576;
  bfloat_t* qkvb   = (bfloat_t*)p; p += 3145728;
  bfloat_t* ctxb   = (bfloat_t*)p; p += 1048576;
  bfloat_t* hhA    = (bfloat_t*)p; p += 1048576;
  bfloat_t* hhB    = (bfloat_t*)p; p += 1048576;
  bfloat_t* mu_b16 = (bfloat_t*)p; p += 1048576;
  bfloat_t* WqkvP  = (bfloat_t*)p; p += 393216;
  bfloat_t* WoP    = (bfloat_t*)p; p += 131072;
  bfloat_t* gnnWP  = (bfloat_t*)p; p += 524288;
  bfloat_t* muWP   = (bfloat_t*)p; p += 131072;
  bfloat_t* lvWP   = (bfloat_t*)p; p += 131072;
  bfloat_t* cpW1aP = (bfloat_t*)p; p += 131072;
  bfloat_t* cpW1cP = (bfloat_t*)p; p += 131072;
  bfloat_t* seW1aP = (bfloat_t*)p; p += 131072;
  bfloat_t* seW1cP = (bfloat_t*)p; p += 131072;
  bfloat_t* pcW1aP = (bfloat_t*)p; p += 131072;
  bfloat_t* pcW1cP = (bfloat_t*)p; p += 131072;
  bfloat_t* cpW2P  = (bfloat_t*)p; p += 65536;

  dim3 blk(256);
  // weight packing (bf16, WMMA B-layout)
  ncd_pack_w_kernel<<<768, blk, 0, stream>>>(Wqkv, WqkvP, 256, 768);
  ncd_pack_w_kernel<<<256, blk, 0, stream>>>(Wo, WoP, 256, 256);
  for (int l = 0; l < 4; ++l)
    ncd_pack_w_kernel<<<256, blk, 0, stream>>>(gnnW + l * 65536, gnnWP + l * 65536, 256, 256);
  ncd_pack_w_kernel<<<256, blk, 0, stream>>>(muW, muWP, 256, 256);
  ncd_pack_w_kernel<<<256, blk, 0, stream>>>(lvW, lvWP, 256, 256);
  ncd_pack_w_kernel<<<256, blk, 0, stream>>>(cpW1,         cpW1aP, 256, 256);
  ncd_pack_w_kernel<<<256, blk, 0, stream>>>(cpW1 + 65536, cpW1cP, 256, 256);
  ncd_pack_w_kernel<<<256, blk, 0, stream>>>(seW1,         seW1aP, 256, 256);
  ncd_pack_w_kernel<<<256, blk, 0, stream>>>(seW1 + 65536, seW1cP, 256, 256);
  ncd_pack_w_kernel<<<256, blk, 0, stream>>>(pcW1,         pcW1aP, 256, 256);
  ncd_pack_w_kernel<<<256, blk, 0, stream>>>(pcW1 + 65536, pcW1cP, 256, 256);
  ncd_pack_w_kernel<<<128, blk, 0, stream>>>(cpW2, cpW2P, 256, 128);

  ncd_prep_kernel<<<2048, blk, 0, stream>>>(vd, emb, h0f, h0b);
  // qkv = h0 @ Wqkv + bqkv : 2048x768x256 -> 1536 wave-tiles -> 192 blocks
  ncd_gemm_bf_kernel<<<192, blk, 0, stream>>>(h0b, WqkvP, bqkv, nullptr, nullptr, qkvb,
                                              2048, 768, 256, 0);
  ncd_attn_kernel<<<BB * NHEADS, blk, 0, stream>>>(qkvb, attn_all, ctxb);
  ncd_attn_mean_kernel<<<1024, blk, 0, stream>>>(attn_all, o_attnw);
  // x = h0 + ctx @ Wo + bo   (2048x256x256 -> 512 wave-tiles -> 64 blocks)
  ncd_gemm_bf_kernel<<<64, blk, 0, stream>>>(ctxb, WoP, bo, h0f, xres, nullptr,
                                             2048, 256, 256, 0);
  ncd_ln_kernel<<<2048, blk, 0, stream>>>(xres, ln_g, ln_b, hhA);
  // 4 GNN layers (bf16 chain, ping-pong)
  ncd_gemm_bf_kernel<<<64, blk, 0, stream>>>(hhA, gnnWP + 0 * 65536, gnnb + 0 * 256, nullptr,
                                             nullptr, hhB, 2048, 256, 256, 1);
  ncd_gemm_bf_kernel<<<64, blk, 0, stream>>>(hhB, gnnWP + 1 * 65536, gnnb + 1 * 256, nullptr,
                                             nullptr, hhA, 2048, 256, 256, 1);
  ncd_gemm_bf_kernel<<<64, blk, 0, stream>>>(hhA, gnnWP + 2 * 65536, gnnb + 2 * 256, nullptr,
                                             nullptr, hhB, 2048, 256, 256, 1);
  ncd_gemm_bf_kernel<<<64, blk, 0, stream>>>(hhB, gnnWP + 3 * 65536, gnnb + 3 * 256, nullptr,
                                             nullptr, hhA, 2048, 256, 256, 1);
  // mu / logvar
  ncd_gemm_bf_kernel<<<64, blk, 0, stream>>>(hhA, muWP, mub, nullptr, o_mu, mu_b16,
                                             2048, 256, 256, 0);
  ncd_gemm_bf_kernel<<<64, blk, 0, stream>>>(hhA, lvWP, lvb, nullptr, o_lv, nullptr,
                                             2048, 256, 256, 0);
  ncd_copy_kernel<<<2048, blk, 0, stream>>>(o_mu, o_hvar, 524288);
  ncd_kl_part_kernel<<<2048, blk, 0, stream>>>(o_mu, o_lv, klp);
  ncd_kl_final_kernel<<<1, blk, 0, stream>>>(klp, o_kl);
  // pairwise projections: a = h_var @ W1[:H] + b1 ; c = h_var @ W1[H:]
  ncd_gemm_bf_kernel<<<64, blk, 0, stream>>>(mu_b16, cpW1aP, cpb1,   nullptr, a_cp, nullptr, 2048, 256, 256, 0);
  ncd_gemm_bf_kernel<<<64, blk, 0, stream>>>(mu_b16, cpW1cP, nullptr, nullptr, c_cp, nullptr, 2048, 256, 256, 0);
  ncd_gemm_bf_kernel<<<64, blk, 0, stream>>>(mu_b16, seW1aP, seb1,   nullptr, a_se, nullptr, 2048, 256, 256, 0);
  ncd_gemm_bf_kernel<<<64, blk, 0, stream>>>(mu_b16, seW1cP, nullptr, nullptr, c_se, nullptr, 2048, 256, 256, 0);
  ncd_gemm_bf_kernel<<<64, blk, 0, stream>>>(mu_b16, pcW1aP, pcb1,   nullptr, a_pc, nullptr, 2048, 256, 256, 0);
  ncd_gemm_bf_kernel<<<64, blk, 0, stream>>>(mu_b16, pcW1cP, nullptr, nullptr, c_pc, nullptr, 2048, 256, 256, 0);
  // fused pairwise scoring (dominant FLOPs, WMMA)
  ncd_pairwise_kernel<<<2048, blk, 0, stream>>>(a_cp, c_cp, a_se, c_se, a_pc, c_pc,
                                                cpW2P, cpb2, cpW3, cpb3,
                                                seW2, seb2, pcW2, pcb2,
                                                o_cp, o_se, o_pc);
}
